// JKNet_31671088840810
// MI455X (gfx1250) — matrix operand order repaired and checked by reading, not compiled, not run on hardware
//
#include <hip/hip_runtime.h>
#include <hip/hip_bf16.h>
#include <stdint.h>

// ---------------------------------------------------------------------------
// JKNet (3x GraphConv + JK-cat + output GEMM) for MI455X / gfx1250.
// Dense GEMMs use v_wmma_f32_16x16x32_bf16 (bf16 in, f32 accumulate).
// Edge aggregation uses edge-parallel global_atomic_add_f32 (L2-resident).
// ---------------------------------------------------------------------------

#define D   128      // feature dim (in == hid == out per layer)
#define D3  384      // JK concat width

typedef __attribute__((ext_vector_type(16))) __bf16 v16bf;
typedef __attribute__((ext_vector_type(8)))  float  v8f;

union Frag16 { uint4 u[2]; v16bf v; };

__device__ __forceinline__ unsigned short f2bf(float f) {
  unsigned u = __float_as_uint(f);
  u += 0x7FFFu + ((u >> 16) & 1u);     // round-to-nearest-even
  return (unsigned short)(u >> 16);
}

// ------------------------------ utility kernels ----------------------------

__global__ void zero_f32(float* __restrict__ p, long n) {
  long i = (long)blockIdx.x * blockDim.x + threadIdx.x;
  if (i < n) p[i] = 0.0f;
}

__global__ void degree_kernel(const int* __restrict__ src, const int* __restrict__ dst,
                              float* __restrict__ outdeg, float* __restrict__ indeg, int E) {
  int i = blockIdx.x * blockDim.x + threadIdx.x;
  if (i < E) {
    atomicAdd(outdeg + src[i], 1.0f);
    atomicAdd(indeg  + dst[i], 1.0f);
  }
}

__global__ void norm_kernel(float* __restrict__ d, int n) {
  int i = blockIdx.x * blockDim.x + threadIdx.x;
  if (i < n) d[i] = rsqrtf(fmaxf(d[i], 1.0f));
}

// xb[n,f] = bf16(feats[n,f] * src_norm[n])
__global__ void prep_x_kernel(const float* __restrict__ x, const float* __restrict__ srcn,
                              unsigned short* __restrict__ xb, long total) {
  long i = (long)blockIdx.x * blockDim.x + threadIdx.x;
  if (i < total) xb[i] = f2bf(x[i] * srcn[i >> 7]);
}

// Swizzle a [K,128] f32 row-major weight matrix into the per-lane WMMA
// B-fragment layout: out[((w*ksteps + t)*32 + lane)*16 + e] where
// k = t*32 + (lane>>4)*16 + e, col = w*16 + (lane&15).  (contiguous k order, same as the A loads)
__global__ void swizzle_w_kernel(const float* __restrict__ W,
                                 unsigned short* __restrict__ out, int K) {
  int idx = blockIdx.x * blockDim.x + threadIdx.x;
  if (idx >= K * D) return;
  const int ksteps = K >> 5;
  int e    = idx & 15;
  int lane = (idx >> 4) & 31;
  int t    = (idx >> 9) % ksteps;
  int w    = idx / (ksteps << 9);
  int k    = t * 32 + (lane >> 4) * 16 + e;
  int col  = w * 16 + (lane & 15);
  out[idx] = f2bf(W[k * D + col]);
}

// ------------------------------- WMMA GEMM ---------------------------------
// C[N,128] = A[N,K](bf16) @ Bsw(K x 128, pre-swizzled bf16) (+ bias)
// One wave computes a 16x16 tile; 8 waves/block cover all 128 output cols.
template <int K>
__global__ __launch_bounds__(256) void wmma_gemm(
    const unsigned short* __restrict__ A,
    const unsigned short* __restrict__ Bsw,
    const float* __restrict__ bias,
    float* __restrict__ C) {
  constexpr int KSTEPS = K / 32;
  const int lane = threadIdx.x & 31;
  const int wave = threadIdx.x >> 5;            // column tile 0..7
  const int half = lane >> 4;
  const int r    = lane & 15;
  const long row = (long)blockIdx.x * 16 + r;   // A row held by this lane
  const unsigned short* aRow = A + row * K;
  const unsigned short* bPtr = Bsw + (size_t)wave * KSTEPS * 512 + (size_t)lane * 16;

  v8f acc = {};
#pragma unroll
  for (int t = 0; t < KSTEPS; ++t) {
    Frag16 a, b;
    // A 16x32 bf16 fragment: elems 0..15 -> K = t*32 + half*16 + e
    const unsigned short* ap = aRow + t * 32 + half * 16;
    a.u[0] = *(const uint4*)(ap);
    a.u[1] = *(const uint4*)(ap + 8);
    // B fragment: 16 contiguous bf16 per lane (pre-swizzled)
    const unsigned short* bp = bPtr + (size_t)t * 512;
    b.u[0] = *(const uint4*)(bp);
    b.u[1] = *(const uint4*)(bp + 8);
    acc = __builtin_amdgcn_wmma_f32_16x16x32_bf16(false, a.v, false, b.v,
                                                  (short)0, acc, false, false);
  }

  // D layout: VGPR v -> row half*8 + v, column lane&15 (ISA 7.12.2)
  const int  col   = wave * 16 + r;
  const float bv   = bias ? bias[col] : 0.0f;
  const long rbase = (long)blockIdx.x * 16 + half * 8;
#pragma unroll
  for (int v = 0; v < 8; ++v)
    C[(rbase + v) * D + col] = acc[v] + bv;
}

// ----------------------------- edge aggregation ----------------------------
// One wave32 per edge; each lane moves float4 chunks. F = 128 or 384.
__global__ __launch_bounds__(256) void scatter_add_kernel(
    const float* __restrict__ h, const int* __restrict__ src,
    const int* __restrict__ dst, float* __restrict__ agg, int E, int F) {
  long gid = (long)blockIdx.x * blockDim.x + threadIdx.x;
  int  e   = (int)(gid >> 5);
  if (e >= E) return;
  int  lane = (int)(gid & 31);
  long bs = (long)src[e] * F;
  long bd = (long)dst[e] * F;
  for (int j = lane * 4; j < F; j += 128) {
    float4 v = *(const float4*)(h + bs + j);
    atomicAdd(agg + bd + j + 0, v.x);
    atomicAdd(agg + bd + j + 1, v.y);
    atomicAdd(agg + bd + j + 2, v.z);
    atomicAdd(agg + bd + j + 3, v.w);
  }
}

// h_l = relu(agg * dst_norm + b); write JK-concat slice (stride 384) and,
// if requested, the next layer's bf16 input bf16(h_l * src_norm).
__global__ void epilogue_kernel(const float* __restrict__ agg,
                                const float* __restrict__ dstn,
                                const float* __restrict__ srcn,
                                const float* __restrict__ bias,
                                float* __restrict__ hj_l,
                                unsigned short* __restrict__ xb_next,
                                long total) {
  long i = (long)blockIdx.x * blockDim.x + threadIdx.x;
  if (i >= total) return;
  long n = i >> 7;
  int  f = (int)(i & 127);
  float v = fmaxf(agg[i] * dstn[n] + bias[f], 0.0f);
  hj_l[n * D3 + f] = v;
  if (xb_next) xb_next[i] = f2bf(v * srcn[n]);
}

__global__ void cvt_bf16_kernel(const float* __restrict__ in,
                                unsigned short* __restrict__ out, long total) {
  long i = (long)blockIdx.x * blockDim.x + threadIdx.x;
  if (i < total) out[i] = f2bf(in[i]);
}

// --------------------------------- driver ----------------------------------

extern "C" void kernel_launch(void* const* d_in, const int* in_sizes, int n_in,
                              void* d_out, int out_size, void* d_ws, size_t ws_size,
                              hipStream_t stream) {
  (void)n_in; (void)out_size; (void)ws_size;

  const float* feats = (const float*)d_in[0];
  const int*   src   = (const int*)d_in[1];
  const int*   dst   = (const int*)d_in[2];
  const float* W[3]  = {(const float*)d_in[3], (const float*)d_in[5], (const float*)d_in[7]};
  const float* B[3]  = {(const float*)d_in[4], (const float*)d_in[6], (const float*)d_in[8]};
  const float* Wout  = (const float*)d_in[9];
  const float* bout  = (const float*)d_in[10];

  const int N = in_sizes[0] / D;   // 50000 (multiple of 16)
  const int E = in_sizes[1];       // 800000

  // -------- workspace carve-up (256B aligned; ~256 MB total) --------
  char*  ws  = (char*)d_ws;
  size_t off = 0;
  auto alloc = [&](size_t bytes) -> char* {
    char* p = ws + off;
    off = (off + bytes + 255) & ~(size_t)255;
    return p;
  };
  float*          srcn  = (float*)alloc((size_t)N * 4);
  float*          dstn  = (float*)alloc((size_t)N * 4);
  unsigned short* wsw[3];
  for (int l = 0; l < 3; ++l) wsw[l] = (unsigned short*)alloc((size_t)D * D * 2);
  unsigned short* wouts = (unsigned short*)alloc((size_t)D3 * D * 2);
  unsigned short* xb    = (unsigned short*)alloc((size_t)N * D * 2);
  float*          hpre  = (float*)alloc((size_t)N * D * 4);
  float*          agg   = (float*)alloc((size_t)N * D * 4);
  float*          hj    = (float*)alloc((size_t)N * D3 * 4);
  float*          agg3  = (float*)alloc((size_t)N * D3 * 4);
  unsigned short* haggb = (unsigned short*)alloc((size_t)N * D3 * 2);

  const dim3 blk(256);
  auto cdiv = [](long a, long b) { return (unsigned)((a + b - 1) / b); };
  const long nd  = (long)N * D;
  const long nd3 = (long)N * D3;
  const unsigned gemmGrid    = (unsigned)(N / 16);           // 3125
  const unsigned scatterGrid = cdiv((long)E * 32, 256);

  // degrees -> rsqrt norms
  zero_f32<<<cdiv(N, 256), blk, 0, stream>>>(srcn, N);
  zero_f32<<<cdiv(N, 256), blk, 0, stream>>>(dstn, N);
  degree_kernel<<<cdiv(E, 256), blk, 0, stream>>>(src, dst, srcn, dstn, E);
  norm_kernel<<<cdiv(N, 256), blk, 0, stream>>>(srcn, N);
  norm_kernel<<<cdiv(N, 256), blk, 0, stream>>>(dstn, N);

  // bf16 inputs + swizzled bf16 weights
  prep_x_kernel<<<cdiv(nd, 256), blk, 0, stream>>>(feats, srcn, xb, nd);
  for (int l = 0; l < 3; ++l)
    swizzle_w_kernel<<<cdiv(D * D, 256), blk, 0, stream>>>(W[l], wsw[l], D);
  swizzle_w_kernel<<<cdiv(D3 * D, 256), blk, 0, stream>>>(Wout, wouts, D3);

  // three GraphConv layers
  for (int l = 0; l < 3; ++l) {
    wmma_gemm<128><<<gemmGrid, blk, 0, stream>>>(xb, wsw[l], nullptr, hpre);
    zero_f32<<<cdiv(nd, 256), blk, 0, stream>>>(agg, nd);
    scatter_add_kernel<<<scatterGrid, blk, 0, stream>>>(hpre, src, dst, agg, E, D);
    epilogue_kernel<<<cdiv(nd, 256), blk, 0, stream>>>(
        agg, dstn, srcn, B[l], hj + (long)l * D, (l < 2) ? xb : nullptr, nd);
  }

  // final un-normalized aggregation over the JK concat, then output GEMM
  zero_f32<<<cdiv(nd3, 256), blk, 0, stream>>>(agg3, nd3);
  scatter_add_kernel<<<scatterGrid, blk, 0, stream>>>(hj, src, dst, agg3, E, D3);
  cvt_bf16_kernel<<<cdiv(nd3, 256), blk, 0, stream>>>(agg3, haggb, nd3);
  wmma_gemm<384><<<gemmGrid, blk, 0, stream>>>(haggb, wouts, bout, (float*)d_out);
}
